// CausalSelfAttention_37117107372360
// MI455X (gfx1250) — matrix operand order repaired and checked by reading, not compiled, not run on hardware
//
#include <hip/hip_runtime.h>

typedef unsigned short u16;
typedef __bf16 v16bf __attribute__((ext_vector_type(16)));
typedef float  v8f   __attribute__((ext_vector_type(8)));
typedef int    v4i   __attribute__((ext_vector_type(4)));

union Frag16 { v16bf bf; uint4 q[2]; u16 us[16]; };

#define BQ 4
#define TT 2048
#define CC 1024
#define HH 16
#define DD 64

#if defined(__has_builtin)
#if __has_builtin(__builtin_amdgcn_global_load_async_to_lds_b128)
#define HAVE_ASYNC 1
#endif
#endif

#if defined(HAVE_ASYNC)
// async global -> LDS b128 copy (ASYNCcnt-tracked, no VGPR staging)
__device__ __forceinline__ void async_b128(const void* g, void* l) {
  __builtin_amdgcn_global_load_async_to_lds_b128(
      (__attribute__((address_space(1))) v4i*)g,
      (__attribute__((address_space(3))) v4i*)l, 0, 0);
}
#define ASYNC_WAIT() asm volatile("s_wait_asynccnt 0x0" ::: "memory")
#else
__device__ __forceinline__ void async_b128(const void* g, void* l) {
  *(uint4*)l = *(const uint4*)g;
}
#define ASYNC_WAIT() ((void)0)
#endif

__device__ __forceinline__ u16 f2bf(float f) {
  unsigned int u = __builtin_bit_cast(unsigned int, f);
  u += 0x7FFFu + ((u >> 16) & 1u);   // round-to-nearest-even
  return (u16)(u >> 16);
}

// reductions across the 16-lane half-groups (masks 1,2,4,8 stay inside a half)
__device__ __forceinline__ float half16_max(float v) {
  v = fmaxf(v, __shfl_xor(v, 1, 32));
  v = fmaxf(v, __shfl_xor(v, 2, 32));
  v = fmaxf(v, __shfl_xor(v, 4, 32));
  v = fmaxf(v, __shfl_xor(v, 8, 32));
  return v;
}
__device__ __forceinline__ float half16_sum(float v) {
  v += __shfl_xor(v, 1, 32);
  v += __shfl_xor(v, 2, 32);
  v += __shfl_xor(v, 4, 32);
  v += __shfl_xor(v, 8, 32);
  return v;
}

__global__ __launch_bounds__(256)
void cvt_bf16(const float* __restrict__ src, u16* __restrict__ dst, int n4) {
  int i = blockIdx.x * blockDim.x + threadIdx.x;
  int stride = gridDim.x * blockDim.x;
  for (; i < n4; i += stride) {
    float4 f = ((const float4*)src)[i];
    ushort4 o;
    o.x = f2bf(f.x); o.y = f2bf(f.y); o.z = f2bf(f.z); o.w = f2bf(f.w);
    ((ushort4*)dst)[i] = o;
  }
}

// C = A(bf16, MxK row-major) * W(bf16, KxN row-major) + bias
// MODE 0: split-scatter qkv as bf16 into [B][H][T][D];  MODE 1: fp32 output.
// Workgroup tile 128x128, 8 waves, each wave a 32x64 tile, K-step 32.
template<int MODE>
__global__ __launch_bounds__(256)
void gemm_bf16(const u16* __restrict__ Am, const u16* __restrict__ Wm,
               const float* __restrict__ bias, int N,
               u16* __restrict__ qd, u16* __restrict__ kd, u16* __restrict__ vd,
               float* __restrict__ outf)
{
  __shared__ u16 at[128][40];   // [row][k], stride 40 halves (80B, 16B aligned)
  __shared__ u16 wt[128][40];   // [n][k] (transposed W tile)

  const int tid  = threadIdx.x;
  const int lane = tid & 31;
  const int w    = tid >> 5;
  const int wm   = w & 3;       // 4 row-blocks of 32
  const int wn   = w >> 2;      // 2 col-blocks of 64
  const int wgM  = blockIdx.y * 128, wgN = blockIdx.x * 128;
  const int g    = lane >> 4, l16 = lane & 15;

  v8f acc[2][4] = {};

  for (int k0 = 0; k0 < CC; k0 += 32) {
    __syncthreads();
    { // stage A tile 128x32 (two async b128 per thread, no transpose needed)
      int row = tid >> 1, kc = (tid & 1) * 16;
      const u16* src = Am + (size_t)(wgM + row) * CC + k0 + kc;
      async_b128(src,     &at[row][kc]);
      async_b128(src + 8, &at[row][kc + 8]);
    }
    { // stage W tile 32x128, transposed into wt[n][k] (VGPR path: transpose)
      int kr = tid >> 3, nc = (tid & 7) * 16;
      const u16* src = Wm + (size_t)(k0 + kr) * N + wgN + nc;
      union { uint4 u; u16 s[8]; } u0, u1;
      u0.u = *(const uint4*)src;
      u1.u = *(const uint4*)(src + 8);
#pragma unroll
      for (int j = 0; j < 8; ++j) { wt[nc + j][kr] = u0.s[j]; wt[nc + 8 + j][kr] = u1.s[j]; }
      if (k0 + 32 < CC) {
        __builtin_prefetch(Am + (size_t)(wgM + (tid >> 1)) * CC + k0 + 32, 0, 0);
        __builtin_prefetch(Wm + (size_t)(k0 + 32 + kr) * N + wgN + nc, 0, 0);
      }
    }
    ASYNC_WAIT();
    __syncthreads();

    // A fragments: lane holds row m; halves 0-15 = k 16g..16g+15
    Frag16 af[2];
#pragma unroll
    for (int mi = 0; mi < 2; ++mi) {
      int m = wm * 32 + mi * 16 + l16;
      af[mi].q[0] = *(const uint4*)&at[m][16 * g];
      af[mi].q[1] = *(const uint4*)&at[m][16 * g + 8];
    }
#pragma unroll
    for (int dg = 0; dg < 4; ++dg) {
      // B fragment: lane = n + 16*(k/16), half = k%16 -> contiguous in wt[n][..]
      Frag16 bf_;
      const u16* wrow = &wt[wn * 64 + dg * 16 + l16][16 * g];
      bf_.q[0] = *(const uint4*)(wrow);
      bf_.q[1] = *(const uint4*)(wrow + 8);
#pragma unroll
      for (int mi = 0; mi < 2; ++mi)
        acc[mi][dg] = __builtin_amdgcn_wmma_f32_16x16x32_bf16(
            false, af[mi].bf, false, bf_.bf, (short)0, acc[mi][dg], false, false);
    }
  }

#pragma unroll
  for (int dg = 0; dg < 4; ++dg) {
    int gn = wgN + wn * 64 + dg * 16 + l16;
    float bv = bias[gn];
#pragma unroll
    for (int mi = 0; mi < 2; ++mi)
#pragma unroll
      for (int r = 0; r < 8; ++r) {
        int gm = wgM + wm * 32 + mi * 16 + 8 * g + r;  // C layout: vgpr r, half-group g
        float v = acc[mi][dg][r] + bv;
        if (MODE == 0) {
          int which = gn >> 10, c = gn & 1023;
          int hh = c >> 6, dd = c & 63;
          int bb = gm >> 11, tt = gm & 2047;
          u16* dst = (which == 0) ? qd : (which == 1) ? kd : vd;
          dst[(((size_t)bb * HH + hh) * TT + tt) * DD + dd] = f2bf(v);
        } else {
          outf[(size_t)gm * CC + gn] = v;
        }
      }
  }
}

// Flash attention: grid (T/128, H, B), 256 threads = 8 waves, 16 q-rows/wave.
__global__ __launch_bounds__(256)
void attn_kernel(const u16* __restrict__ qg, const u16* __restrict__ kg,
                 const u16* __restrict__ vg, u16* __restrict__ ctx)
{
  __shared__ u16 kt[32][72];        // [key][d] row-major
  __shared__ u16 vt[64][40];        // [d][key] transposed
  __shared__ u16 pbuf[8][16][40];   // per-wave P scratch for C->A relayout

  const int tid  = threadIdx.x;
  const int lane = tid & 31;
  const int w    = tid >> 5;
  const int b    = blockIdx.z, h = blockIdx.y;
  const int wg_q0 = blockIdx.x * 128;
  const int q0   = wg_q0 + w * 16;
  const int g    = lane >> 4;
  const int l16  = lane & 15;
  const int mbase = 8 * g;

  const size_t head = ((size_t)b * HH + h) * (size_t)TT * DD;
  const u16* qp = qg + head;
  const u16* kp = kg + head;
  const u16* vp = vg + head;

  // Q fragments (A layout) for D split into two 32-wide chunks, loaded once.
  Frag16 qf[2];
  {
    const u16* qrow = qp + (size_t)(q0 + l16) * DD;
#pragma unroll
    for (int c = 0; c < 2; ++c) {
      qf[c].q[0] = *(const uint4*)(qrow + 32 * c + 16 * g);
      qf[c].q[1] = *(const uint4*)(qrow + 32 * c + 16 * g + 8);
    }
  }

  v8f acc[4] = {};
  float mrow[8], lrow[8];
#pragma unroll
  for (int r = 0; r < 8; ++r) { mrow[r] = -__builtin_inff(); lrow[r] = 0.f; }

  const int kmax = wg_q0 + 128;     // workgroup-uniform causal bound
  for (int kb = 0; kb < kmax; kb += 32) {
    __syncthreads();
    { // cooperative stage: K row-major (async DMA), V transposed (VGPR path)
      int key = tid >> 3;
      int dc  = (tid & 7) * 8;
      async_b128(kp + (size_t)(kb + key) * DD + dc, &kt[key][dc]);
      union { uint4 u; u16 s[8]; } vv;
      vv.u = *(const uint4*)(vp + (size_t)(kb + key) * DD + dc);
#pragma unroll
      for (int j = 0; j < 8; ++j) vt[dc + j][key] = vv.s[j];
      if (kb + 32 < kmax) {
        __builtin_prefetch(kp + (size_t)(kb + 32 + key) * DD + dc, 0, 0);
        __builtin_prefetch(vp + (size_t)(kb + 32 + key) * DD + dc, 0, 0);
      }
    }
    ASYNC_WAIT();
    __syncthreads();

    if (kb > q0 + 15) continue;     // wave-uniform: EXEC stays all-ones

    // S = Q * K^T, two 16-key tiles, K-dim = D (2 chunks of 32)
    v8f st[2];
#pragma unroll
    for (int j = 0; j < 2; ++j) {
      v8f s = {};
#pragma unroll
      for (int c = 0; c < 2; ++c) {
        Frag16 bfr;
        const u16* krow = &kt[16 * j + l16][32 * c + 16 * g];
        bfr.q[0] = *(const uint4*)(krow);
        bfr.q[1] = *(const uint4*)(krow + 8);
        s = __builtin_amdgcn_wmma_f32_16x16x32_bf16(
            false, qf[c].bf, false, bfr.bf, (short)0, s, false, false);
      }
      int key = kb + 16 * j + l16;
#pragma unroll
      for (int r = 0; r < 8; ++r) {
        float sv = s[r] * 0.125f;                         // 1/sqrt(64)
        if (key > q0 + mbase + r) sv = -__builtin_inff(); // causal mask
        s[r] = sv;
      }
      st[j] = s;
    }

    // online softmax update
    float mnew[8], alpha[8], psum[8];
#pragma unroll
    for (int r = 0; r < 8; ++r) {
      float sm = fmaxf(st[0][r], st[1][r]);
      sm = half16_max(sm);
      mnew[r] = fmaxf(mrow[r], sm);
      alpha[r] = __expf(mrow[r] - mnew[r]);
      psum[r] = 0.f;
    }
#pragma unroll
    for (int j = 0; j < 2; ++j)
#pragma unroll
      for (int r = 0; r < 8; ++r) {
        float p = __expf(st[j][r] - mnew[r]);
        psum[r] += p;
        pbuf[w][mbase + r][16 * j + l16] = f2bf(p);       // C-layout -> LDS
      }
#pragma unroll
    for (int r = 0; r < 8; ++r) {
      lrow[r] = lrow[r] * alpha[r] + half16_sum(psum[r]);
      mrow[r] = mnew[r];
    }
#pragma unroll
    for (int dg = 0; dg < 4; ++dg)
#pragma unroll
      for (int r = 0; r < 8; ++r) acc[dg][r] *= alpha[r];

    // reload P in A layout (same-wave DS ops are in-order)
    Frag16 pf;
    pf.q[0] = *(const uint4*)&pbuf[w][l16][16 * g];
    pf.q[1] = *(const uint4*)&pbuf[w][l16][16 * g + 8];

#pragma unroll
    for (int dg = 0; dg < 4; ++dg) {
      Frag16 vfr;
      const u16* vrow = &vt[dg * 16 + l16][16 * g];
      vfr.q[0] = *(const uint4*)(vrow);
      vfr.q[1] = *(const uint4*)(vrow + 8);
      acc[dg] = __builtin_amdgcn_wmma_f32_16x16x32_bf16(
          false, pf.bf, false, vfr.bf, (short)0, acc[dg], false, false);
    }
  }

  // normalize and write ctx as bf16 in [B][T][H*D]
#pragma unroll
  for (int r = 0; r < 8; ++r) {
    float inv = 1.f / lrow[r];
    int row = q0 + mbase + r;
    size_t base = ((size_t)b * TT + row) * CC + h * DD;
#pragma unroll
    for (int dg = 0; dg < 4; ++dg)
      ctx[base + dg * 16 + l16] = f2bf(acc[dg][r] * inv);
  }
}

extern "C" void kernel_launch(void* const* d_in, const int* in_sizes, int n_in,
                              void* d_out, int out_size, void* d_ws, size_t ws_size,
                              hipStream_t stream) {
  const float* x     = (const float*)d_in[0];
  const float* w_qkv = (const float*)d_in[1];
  const float* b_qkv = (const float*)d_in[2];
  const float* w_out = (const float*)d_in[3];
  const float* b_out = (const float*)d_in[4];
  float* out = (float*)d_out;

  const size_t NX    = (size_t)BQ * TT * CC;   // 8,388,608
  const size_t NWQKV = (size_t)CC * 3 * CC;    // 3,145,728
  const size_t NWO   = (size_t)CC * CC;        // 1,048,576

  u16* xb    = (u16*)d_ws;
  u16* wqkvb = xb + NX;
  u16* wob   = wqkvb + NWQKV;
  u16* q_ws  = wob + NWO;
  u16* k_ws  = q_ws + NX;
  u16* v_ws  = k_ws + NX;
  u16* ctx   = v_ws + NX;

  cvt_bf16<<<2048, 256, 0, stream>>>(x,     xb,    (int)(NX / 4));
  cvt_bf16<<<1024, 256, 0, stream>>>(w_qkv, wqkvb, (int)(NWQKV / 4));
  cvt_bf16<<<512,  256, 0, stream>>>(w_out, wob,   (int)(NWO / 4));

  gemm_bf16<0><<<dim3(3 * CC / 128, BQ * TT / 128), 256, 0, stream>>>(
      xb, wqkvb, b_qkv, 3 * CC, q_ws, k_ws, v_ws, nullptr);

  attn_kernel<<<dim3(TT / 128, HH, BQ), 256, 0, stream>>>(q_ws, k_ws, v_ws, ctx);

  gemm_bf16<1><<<dim3(CC / 128, BQ * TT / 128), 256, 0, stream>>>(
      ctx, wob, b_out, CC, nullptr, nullptr, nullptr, out);
}
